// ModelClass_48284022342134
// MI455X (gfx1250) — matrix operand order repaired and checked by reading, not compiled, not run on hardware
//
#include <hip/hip_runtime.h>
#include <hip/hip_bf16.h>
#include <stdint.h>

// ---------------- types for WMMA ----------------
typedef __attribute__((ext_vector_type(16))) __bf16 v16bf;
typedef __attribute__((ext_vector_type(8)))  float  v8f;

__device__ __host__ __forceinline__ unsigned short f2bf_host(float x) {
    union { float f; unsigned u; } c; c.f = x;
    unsigned u = c.u;
    return (unsigned short)((u + 0x7FFFu + ((u >> 16) & 1u)) >> 16);
}
__device__ __forceinline__ unsigned short f2bf(float x) {
    union { float f; unsigned u; } c; c.f = x;
    unsigned u = c.u;
    return (unsigned short)((u + 0x7FFFu + ((u >> 16) & 1u)) >> 16);   // RNE
}

// branchless guarded load: clamp address, select zero (no exec-mask branches)
__device__ __forceinline__ float ldg0(const float* __restrict__ p, size_t idx, bool ok) {
    float v = p[ok ? idx : 0];
    return ok ? v : 0.0f;
}

#define LDST 40   // LDS row stride in halves (80B = 5*16B: aligned, bank-spread)

// ---------------------------------------------------------------------------
// Weight pre-pass: Wt[col][k] = bf16(W[k][col]), zero-padded to (Npad, Kpad).
// Runs once per weight per launch; weights are then streamed to LDS with
// async b128 copies inside the GEMM (no per-tile conversion).
// ---------------------------------------------------------------------------
__global__ void convW(const float* __restrict__ W, unsigned short* __restrict__ Wt,
                      int K, int N, int Kpad, int Npad) {
    int tid = blockIdx.x * blockDim.x + threadIdx.x;
    if (tid >= Npad * Kpad) return;
    int k = tid % Kpad, col = tid / Kpad;
    bool ok = (k < K) && (col < N);
    float v = ok ? W[(size_t)k * N + col] : 0.0f;
    Wt[tid] = f2bf(v);
}

// ---------------------------------------------------------------------------
// Fused GEMM: C[M,N] = act(A[M,K] @ W[K,N] + bias) (+ residual), leaky 0.1.
// Block = 128 threads = 4 waves covering 64 rows x N (N <= NT*16).
// Weight tile (bf16, pre-transposed) streamed into LDS with
// GLOBAL_LOAD_ASYNC_TO_LDS_B128 (ASYNCcnt), overlapped with the A-tile
// fp32->bf16 staging; each wave issues NT independent v_wmma_f32_16x16x32_bf16.
// ---------------------------------------------------------------------------
template <int NT>
__global__ __launch_bounds__(128)
void wmma_gemm(const float* __restrict__ A, const unsigned short* __restrict__ Wt,
               const float* __restrict__ bias, const float* resid,
               float* C, int M, int K, int Kpad, int N, int act) {
    __shared__ alignas(16) unsigned short As[4][16 * LDST];   // per-wave A tile
    __shared__ alignas(16) unsigned short Ws[NT * 16 * LDST]; // W tile [col][k]

    const int tid  = threadIdx.x;
    const int wv   = tid >> 5;
    const int lane = tid & 31;
    const int m0   = (blockIdx.x * 4 + wv) * 16;

    union Frag { v16bf v; unsigned short s[16]; } af, bf;
    v8f acc[NT];
#pragma unroll
    for (int t = 0; t < NT; ++t) acc[t] = v8f{0.f,0.f,0.f,0.f,0.f,0.f,0.f,0.f};

    for (int k0 = 0; k0 < K; k0 += 32) {
        // ---- async-stage W tile: 4 x 16B per column, global(bf16) -> LDS ----
        {
            constexpr int UNITS = NT * 16 * 4;        // 16-byte units
            for (int u = tid; u < UNITS; u += 128) {
                int col = u >> 2, part = u & 3;
                unsigned ldsoff = (unsigned)(uintptr_t)&Ws[col * LDST + part * 8];
                const unsigned short* gp = Wt + (size_t)col * Kpad + k0 + part * 8;
                asm volatile("global_load_async_to_lds_b128 %0, %1, off"
                             :: "v"(ldsoff), "v"(gp) : "memory");
            }
        }
        // ---- stage this wave's A tile (fp32 -> bf16), overlapped ----
        for (int i = 0; i < 16; ++i) {
            int grow = m0 + i, gk = k0 + lane;
            bool ok = (grow < M) && (gk < K);
            As[wv][i * LDST + lane] = f2bf(ldg0(A, (size_t)grow * K + gk, ok));
        }
        asm volatile("s_wait_asynccnt 0x0" ::: "memory");
        __syncthreads();

        // ---- A fragment (16x32 bf16, contiguous k = 16h + s, same order as B) ----
        {
            int arow  = lane & 15;
            int kbase = (lane >> 4) * 16;
            const unsigned short* ap = &As[wv][arow * LDST];
#pragma unroll
            for (int i = 0; i < 8; ++i) af.s[i] = ap[kbase + i];          // K kbase..+7
#pragma unroll
            for (int i = 0; i < 8; ++i) af.s[8 + i] = ap[kbase + 8 + i];  // K kbase+8..+15
        }
        // ---- per n-tile: contiguous B fragment + WMMA ----
        {
            int bcol = lane & 15;
            int kb2  = (lane >> 4) * 16;
#pragma unroll
            for (int t = 0; t < NT; ++t) {
                const unsigned short* bp = &Ws[(t * 16 + bcol) * LDST + kb2];
#pragma unroll
                for (int i = 0; i < 16; ++i) bf.s[i] = bp[i];
                acc[t] = __builtin_amdgcn_wmma_f32_16x16x32_bf16(
                             false, af.v, false, bf.v, (short)0, acc[t], false, false);
            }
        }
        __syncthreads();
    }

    // ---- epilogue: lane owns col = lane&15 of each n-tile, rows rbase..+7 ----
    int col   = lane & 15;
    int rbase = (lane >> 4) * 8;
#pragma unroll
    for (int t = 0; t < NT; ++t) {
        int gc = t * 16 + col;
        if (gc < N) {
            float bv = bias ? bias[gc] : 0.0f;
#pragma unroll
            for (int r = 0; r < 8; ++r) {
                int grow = m0 + rbase + r;
                if (grow < M) {
                    float v = acc[t][r] + bv;
                    if (act) v = v > 0.0f ? v : 0.1f * v;
                    if (resid) v += resid[(size_t)grow * N + gc];
                    C[(size_t)grow * N + gc] = v;
                }
            }
        }
    }
}

// ---------------------------------------------------------------------------
// CNU stats: S = sum_n h, W = mean_n |h - S/div|; write [S|W] rows of width 2L
// ---------------------------------------------------------------------------
__global__ void cnu_stats(const float* __restrict__ H, float* gin,
                          int n, int L, float div) {
    int b = blockIdx.x, l = threadIdx.x;           // blockDim = L
    const float* hp = H + (size_t)b * n * L + l;
    float s = 0.f;
    for (int i = 0; i < n; ++i) s += hp[(size_t)i * L];
    float m = s / div, w = 0.f;
    for (int i = 0; i < n; ++i) w += fabsf(hp[(size_t)i * L] - m);
    gin[(size_t)b * 2 * L + l]     = s;
    gin[(size_t)b * 2 * L + L + l] = w / (float)n;
}

// concat per-node h (rows,L) with per-graph g (B,G) broadcast -> (rows, L+G)
__global__ void concat_hg(const float* __restrict__ h, const float* __restrict__ g,
                          float* out, int nloc, int L, int G, int B) {
    int tid = blockIdx.x * blockDim.x + threadIdx.x;
    int tot = B * nloc * (L + G);
    if (tid >= tot) return;
    int j = tid % (L + G);
    int row = tid / (L + G);
    int b = row / nloc;
    out[tid] = (j < L) ? h[(size_t)row * L + j] : g[(size_t)b * G + (j - L)];
}

// per-graph feature sum over nodes, concatenated with condition (width f+3)
__global__ void hsum_concat(const float* __restrict__ h, const float* __restrict__ cond,
                            float* out, int n, int f) {
    int b = blockIdx.x, j = threadIdx.x;           // blockDim = f+3
    if (j < f) {
        float s = 0.f;
        const float* hp = h + (size_t)b * n * f + j;
        for (int i = 0; i < n; ++i) s += hp[(size_t)i * f];
        out[b * (f + 3) + j] = s;
    } else {
        out[b * (f + 3) + j] = cond[b * 3 + (j - f)];
    }
}

// global sum & max pools over axis 1
__global__ void lat_pool(const float* __restrict__ x, float* osum, float* omax,
                         int n, int f) {
    int b = blockIdx.x, j = threadIdx.x;           // blockDim = f
    const float* xp = x + (size_t)b * n * f + j;
    float s = 0.f, mx = -3.402823466e38f;
    for (int i = 0; i < n; ++i) { float v = xp[(size_t)i * f]; s += v; mx = fmaxf(mx, v); }
    osum[b * f + j] = s;
    omax[b * f + j] = mx;
}

// GATv2 attention: per (b, centroid r, head): logits over n sources, softmax,
// weighted sum of source messages gl. gl:(B*n,H*f), gr:(r,H*f), att:(H,f).
__global__ __launch_bounds__(64)
void gat_attn(const float* __restrict__ gl, const float* __restrict__ gr,
              const float* __restrict__ att, float* out, int n, int r, int H, int f) {
    __shared__ float lg[1024];
    __shared__ float red[64];
    __shared__ float grs[64];
    __shared__ float ats[64];
    int blk = blockIdx.x;
    int hd = blk % H, rr = (blk / H) % r, b = blk / (H * r);
    int t = threadIdx.x;                           // blockDim = f = 64
    grs[t] = gr[(size_t)rr * (H * f) + hd * f + t];
    ats[t] = att[hd * f + t];
    __syncthreads();
    for (int i = t; i < n; i += 64) {
        const float* glp = gl + ((size_t)b * n + i) * (H * f) + hd * f;
        float acc = 0.f;
        for (int d = 0; d < f; ++d) {
            float e = glp[d] + grs[d];
            e = e > 0.f ? e : 0.2f * e;            // GATv2 leaky 0.2
            acc += e * ats[d];
        }
        lg[i] = acc;
    }
    __syncthreads();
    float mx = -3.402823466e38f;
    for (int i = t; i < n; i += 64) mx = fmaxf(mx, lg[i]);
    red[t] = mx; __syncthreads();
    for (int s = 32; s > 0; s >>= 1) { if (t < s) red[t] = fmaxf(red[t], red[t + s]); __syncthreads(); }
    mx = red[0]; __syncthreads();
    float se = 0.f;
    for (int i = t; i < n; i += 64) { float e = __expf(lg[i] - mx); lg[i] = e; se += e; }
    red[t] = se; __syncthreads();
    for (int s = 32; s > 0; s >>= 1) { if (t < s) red[t] += red[t + s]; __syncthreads(); }
    float denom = red[0];
    __syncthreads();
    float acc = 0.f;                               // thread t = output dim d
    for (int i = 0; i < n; ++i)
        acc += lg[i] * gl[((size_t)b * n + i) * (H * f) + hd * f + t];
    out[((size_t)(b * r + rr) * H + hd) * f + t] = acc / denom;
}

// mean over heads + out bias
__global__ void pool_combine(const float* __restrict__ praw, const float* __restrict__ bias,
                             float* xout, int B, int r, int f, int H) {
    int tid = blockIdx.x * blockDim.x + threadIdx.x;
    if (tid >= B * r * f) return;
    int d = tid % f;
    int rf = tid / f;                              // b*r + rr
    float s = 0.f;
    for (int h = 0; h < H; ++h) s += praw[((size_t)rf * H + h) * f + d];
    xout[tid] = s / (float)H + bias[d];
}

// ---------------------------------------------------------------------------
extern "C" void kernel_launch(void* const* d_in, const int* in_sizes, int n_in,
                              void* d_out, int out_size, void* d_ws, size_t ws_size,
                              hipStream_t stream) {
    const float* X    = (const float*)d_in[0];     // (32,1024,4)
    const float* COND = (const float*)d_in[1];     // (32,3)

    // -------- parameter pytree (jax flatten: dict keys sorted) --------
    int idx = 2;
    auto nxt = [&]() -> const float* { return (const float*)d_in[idx++]; };
    struct FFN  { const float* W[3]; const float* b[3]; };
    struct CNUP { FFN emb, glob, out; };
    auto takeFFN = [&](FFN& f)  { for (int i = 0; i < 3; ++i) { f.W[i] = nxt(); f.b[i] = nxt(); } };
    auto takeCNU = [&](CNUP& c) { takeFFN(c.emb); takeFFN(c.glob); takeFFN(c.out); };

    CNUP dcnu[3][2]; FFN dffn[3];                  // "disc" first
    for (int l = 0; l < 3; ++l) { takeCNU(dcnu[l][0]); takeCNU(dcnu[l][1]); takeFFN(dffn[l]); }
    CNUP ecnu[2]; FFN einp[2];                     // then "emb" ("cnu" < "inp")
    for (int l = 0; l < 2; ++l) { takeCNU(ecnu[l]); takeFFN(einp[l]); }
    struct PoolP { const float *Wl, *Wr, *att, *bias, *bl, *br, *cent; };
    PoolP pp[2];                                   // then "pool": Wl,Wr,att,bias,bl,br,cent
    for (int l = 0; l < 2; ++l) {
        pp[l].Wl = nxt(); pp[l].Wr = nxt(); pp[l].att = nxt(); pp[l].bias = nxt();
        pp[l].bl = nxt(); pp[l].br = nxt(); pp[l].cent = nxt();
    }

    // -------- workspace layout (floats) --------
    float* ws   = (float*)d_ws;
    float* bufP = ws;                 // 4194304  (32768 x 128)
    float* bufQ = ws + 4194304;       // 4194304
    float* bufR = ws + 8388608;       // 2097152  (32768 x 64)
    float* bufH = ws + 10485760;      // 2097152
    float* bufX = ws + 12582912;      // 32768    (pooled x)
    float* gin  = ws + 12615680;      // 4096     (B,128)
    float* gA   = ws + 12619776;      // 4096
    float* gB   = ws + 12623872;      // 4096
    float* gO   = ws + 12627968;      // 2048     (B,64)
    float* dinp = ws + 12630016;      // 2240     (B, f+3)
    float* grb  = ws + 12632256;      // 2048     (r, 128)
    float* praw = ws + 12634304;      // 65536    (B, r, 2, 64)
    // bf16 transposed weight slots: 91 GEMMs x 128x128 halves (32KB) each
    unsigned short* wsBf = (unsigned short*)(ws + 12699840);
    int wslot = 0;

    auto gemm = [&](const float* A, const float* W, const float* b, const float* res,
                    float* C, int M, int K, int N, bool act) {
        int ntD  = (N <= 16) ? 1 : (N <= 32) ? 2 : (N <= 64) ? 4 : 8;
        int Kpad = ((K + 31) / 32) * 32;
        int Npad = ntD * 16;
        unsigned short* Wt = wsBf + (size_t)wslot * 16384;
        ++wslot;
        convW<<<(Npad * Kpad + 255) / 256, 256, 0, stream>>>(W, Wt, K, N, Kpad, Npad);
        dim3 g((unsigned)((M + 63) / 64));
        dim3 blk(128);
        int a = act ? 1 : 0;
        if (ntD == 1)      wmma_gemm<1><<<g, blk, 0, stream>>>(A, Wt, b, res, C, M, K, Kpad, N, a);
        else if (ntD == 2) wmma_gemm<2><<<g, blk, 0, stream>>>(A, Wt, b, res, C, M, K, Kpad, N, a);
        else if (ntD == 4) wmma_gemm<4><<<g, blk, 0, stream>>>(A, Wt, b, res, C, M, K, Kpad, N, a);
        else               wmma_gemm<8><<<g, blk, 0, stream>>>(A, Wt, b, res, C, M, K, Kpad, N, a);
    };

    const int B = 32;
    const int FE[3]   = {4, 64, 64};
    const int RR[2]   = {16, 4};
    const int sumOff[3] = {96, 352, 4448};
    const int maxOff[3] = {224, 2400, 6496};
    float* fout = (float*)d_out;

    // residual CNU block, in-place on h (rows = B*nloc, width f)
    auto cnuUpdate = [&](const CNUP& c, float* h, int rows, int f, int G,
                         float div, int nloc) {
        gemm(h,    c.emb.W[0], c.emb.b[0], nullptr, bufP, rows, f,   128, true);
        gemm(bufP, c.emb.W[1], c.emb.b[1], nullptr, bufQ, rows, 128, 128, true);
        gemm(bufQ, c.emb.W[2], c.emb.b[2], nullptr, bufR, rows, 128, 64,  true);
        cnu_stats<<<B, 64, 0, stream>>>(bufR, gin, nloc, 64, div);
        gemm(gin, c.glob.W[0], c.glob.b[0], nullptr, gA, B, 128, 128, true);
        gemm(gA,  c.glob.W[1], c.glob.b[1], nullptr, gB, B, 128, 128, true);
        gemm(gB,  c.glob.W[2], c.glob.b[2], nullptr, gO, B, 128, G,   true);
        int tot = rows * (64 + G);
        concat_hg<<<(tot + 255) / 256, 256, 0, stream>>>(bufR, gO, bufP, nloc, 64, G, B);
        gemm(bufP, c.out.W[0], c.out.b[0], nullptr, bufQ, rows, 64 + G, 128, true);
        gemm(bufQ, c.out.W[1], c.out.b[1], nullptr, bufP, rows, 128,    128, true);
        gemm(bufP, c.out.W[2], c.out.b[2], h,       h,    rows, 128,    f,   false);
    };

    const float* xcur = X;
    int n = 1024;
    for (int l = 0; l < 3; ++l) {
        int f = FE[l];
        int rows = B * n;
        float div = (l == 0) ? (float)n : (float)B;

        // global add / max pools -> outputs
        lat_pool<<<B, f, 0, stream>>>(xcur, fout + sumOff[l], fout + maxOff[l], n, f);

        // critic CNU stack (h = copy of x, updated in place)
        hipMemcpyAsync(bufH, xcur, sizeof(float) * (size_t)rows * f,
                       hipMemcpyDeviceToDevice, stream);
        cnuUpdate(dcnu[l][0], bufH, rows, f, 32, div, n);
        cnuUpdate(dcnu[l][1], bufH, rows, f, 32, div, n);

        // discriminator head -> crit rows [l*B, l*B+32)
        hsum_concat<<<B, f + 3, 0, stream>>>(bufH, COND, dinp, n, f);
        gemm(dinp, dffn[l].W[0], dffn[l].b[0], nullptr, gA, B, f + 3, 128, true);
        gemm(gA,   dffn[l].W[1], dffn[l].b[1], nullptr, gB, B, 128,   128, true);
        gemm(gB,   dffn[l].W[2], dffn[l].b[2], nullptr, fout + l * B, B, 128, 1, false);
        if (l == 2) break;

        // embedding: h2 = FFN_inp(x); x = CNU(h2) + h2
        int f2 = FE[l + 1];
        gemm(xcur, einp[l].W[0], einp[l].b[0], nullptr, bufP, rows, f,   128, true);
        gemm(bufP, einp[l].W[1], einp[l].b[1], nullptr, bufQ, rows, 128, 128, true);
        gemm(bufQ, einp[l].W[2], einp[l].b[2], nullptr, bufH, rows, 128, f2,  false);
        cnuUpdate(ecnu[l], bufH, rows, f2, 64, div, n);

        // bipartite GATv2 pool to r centroids
        int r = RR[l];
        gemm(bufH,       pp[l].Wl, pp[l].bl, nullptr, bufP, rows, f2, 2 * f2, false);
        gemm(pp[l].cent, pp[l].Wr, pp[l].br, nullptr, grb,  r,    f2, 2 * f2, false);
        gat_attn<<<B * r * 2, 64, 0, stream>>>(bufP, grb, pp[l].att, praw, n, r, 2, f2);
        int tot = B * r * f2;
        pool_combine<<<(tot + 255) / 256, 256, 0, stream>>>(praw, pp[l].bias, bufX, B, r, f2, 2);
        xcur = bufX;
        n = r;
    }
}
